// SpatialGather_48026324304270
// MI455X (gfx1250) — matrix operand order repaired and checked by reading, not compiled, not run on hardware
//
#include <hip/hip_runtime.h>
#include <hip/hip_bf16.h>

typedef __bf16  bf16_t;
typedef bf16_t  v16bf __attribute__((ext_vector_type(16)));
typedef float   v8f   __attribute__((ext_vector_type(8)));
typedef unsigned u32x4 __attribute__((ext_vector_type(4)));
typedef float    f32x4 __attribute__((ext_vector_type(4)));

#define NPIX   (512*512)     // H*W
#define NCH    512           // C
#define NCLS   150           // K
#define KB     32            // classes per block (2 k-tiles)
#define KT2    5             // 160/32 blocks along k
#define LDR    513           // R row stride: 512 channels + 1 "sum" column
#define CHUNK  1024          // pixels per block
#define NCHUNK (NPIX/CHUNK)  // 256
#define STEP   32            // WMMA K per step

// rounded f32 -> bf16 (used only in small A staging)
__device__ __forceinline__ unsigned short f2bf(float x){
    unsigned u = __builtin_bit_cast(unsigned, x);
    return (unsigned short)((u + 0x8000u) >> 16);
}
// single v_perm_b32: pack two f32 (truncate) into packed bf16 pair {hi,lo}
__device__ __forceinline__ unsigned pk2bf(float lo, float hi){
    return __builtin_amdgcn_perm(__builtin_bit_cast(unsigned, hi),
                                 __builtin_bit_cast(unsigned, lo),
                                 0x07060302u);
}

union AFrag { u32x4 q[2]; v16bf v; };
union BFrag { unsigned u[8]; v16bf v; };

__global__ void zeroR(float* __restrict__ R){
    int i = blockIdx.x*blockDim.x + threadIdx.x;
    if (i < 160*LDR) R[i] = 0.f;
}

// per-pixel online softmax stats: m[n] = max_k, r[n] = 1/sum exp(l-m)
__global__ __launch_bounds__(256)
void softmax_stats(const float* __restrict__ logits,
                   float* __restrict__ m_out, float* __restrict__ r_out){
    int n = blockIdx.x*blockDim.x + threadIdx.x;
    if (n >= NPIX) return;
    float m = -3.0e38f, Z = 0.f;
    for (int k = 0; k < NCLS; ++k){
        float x = logits[(size_t)k*NPIX + n];
        float mn = fmaxf(m, x);
        Z = Z*__expf(m - mn) + __expf(x - mn);
        m = mn;
    }
    m_out[n] = m;
    r_out[n] = 1.f / Z;
}

// Block: 32 classes (blockIdx.x) x all 512 channels x CHUNK pixels (blockIdx.y).
// 8 waves; wave w owns c-tiles 4w..4w+3 for BOTH k-halves (B converted once, used twice).
// Wave 0 additionally runs two ones-column WMMAs to accumulate S[k].
__global__ __launch_bounds__(256)
void gemm_wmma(const float* __restrict__ feats, const float* __restrict__ logits,
               const float* __restrict__ sm_m, const float* __restrict__ sm_r,
               float* __restrict__ R){
    __shared__ __attribute__((aligned(16))) unsigned short pA[2][KB*STEP]; // bf16 probs [class][pixel]

    const int tid   = threadIdx.x;
    const int lane  = tid & 31;
    const int wv    = tid >> 5;       // 0..7
    const int half  = lane >> 4;      // 0/1
    const int mcol  = lane & 15;      // row-of-A / col-of-B index
    const int k0    = blockIdx.x * KB;
    const int nbase = blockIdx.y * CHUNK;

    v8f acc[8] = {};                  // [kt*4 + t]
    v8f accS[2] = {};
    const int ct0  = wv * 4;
    const int scls = tid >> 5;        // staging class (+8 per round)
    const int spix = tid & 31;        // staging pixel within step

    for (int s = 0; s < CHUNK/STEP; ++s){
        const int n0  = nbase + s*STEP;
        const int buf = s & 1;

        // ---- stage A: p = exp(l - m) * rinv, bf16 into LDS (32 classes x 32 px) ----
        {
            float mm = sm_m[n0 + spix];
            float rr = sm_r[n0 + spix];
            #pragma unroll
            for (int rnd = 0; rnd < 4; ++rnd){
                int cls = scls + rnd*8;
                int kk  = k0 + cls;
                float l = (kk < NCLS) ? logits[(size_t)kk*NPIX + n0 + spix] : -3.0e38f;
                float p = __expf(l - mm) * rr;
                pA[buf][cls*STEP + spix] = f2bf(p);
            }
        }
        __syncthreads();

        // ---- two A fragments (16-bit 16x32 layout, §7.12.2) ----
        AFrag a0, a1;
        const char* abase = (const char*)(&pA[buf][0]);
        const char* ab0 = abase + mcol*64 + half*16;
        const char* ab1 = abase + (16 + mcol)*64 + half*16;
        a0.q[0] = *(const u32x4*)(ab0);
        a0.q[1] = *(const u32x4*)(ab0 + 32);
        a1.q[0] = *(const u32x4*)(ab1);
        a1.q[1] = *(const u32x4*)(ab1 + 32);

        // ---- B fragments from feats; each feeds both k-half WMMAs ----
        #pragma unroll
        for (int t = 0; t < 4; ++t){
            int ch = (ct0 + t)*16 + mcol;
            const f32x4* fp = (const f32x4*)(feats + (size_t)ch*NPIX + n0 + half*8);
            f32x4 f0 = fp[0], f1 = fp[1], f2 = fp[4], f3 = fp[5];
            BFrag b;
            b.u[0]=pk2bf(f0.x,f0.y); b.u[1]=pk2bf(f0.z,f0.w);
            b.u[2]=pk2bf(f1.x,f1.y); b.u[3]=pk2bf(f1.z,f1.w);
            b.u[4]=pk2bf(f2.x,f2.y); b.u[5]=pk2bf(f2.z,f2.w);
            b.u[6]=pk2bf(f3.x,f3.y); b.u[7]=pk2bf(f3.z,f3.w);
            acc[t]   = __builtin_amdgcn_wmma_f32_16x16x32_bf16(
                          false, a0.v, false, b.v, (short)0, acc[t],   false, false);
            acc[4+t] = __builtin_amdgcn_wmma_f32_16x16x32_bf16(
                          false, a1.v, false, b.v, (short)0, acc[4+t], false, false);
        }
        // ---- wave 0: ones-column B tile -> partial S[k] in column 0 ----
        if (wv == 0){
            BFrag b;
            unsigned one2 = (mcol == 0) ? 0x3F803F80u : 0u; // bf16 1.0 pair
            #pragma unroll
            for (int i = 0; i < 8; ++i) b.u[i] = one2;
            accS[0] = __builtin_amdgcn_wmma_f32_16x16x32_bf16(
                          false, a0.v, false, b.v, (short)0, accS[0], false, false);
            accS[1] = __builtin_amdgcn_wmma_f32_16x16x32_bf16(
                          false, a1.v, false, b.v, (short)0, accS[1], false, false);
        }
    }

    // ---- accumulate D tiles into global R (D layout: VGPR r -> M=r+half*8, lane -> N) ----
    #pragma unroll
    for (int kt = 0; kt < 2; ++kt){
        #pragma unroll
        for (int t = 0; t < 4; ++t){
            int cbase = (ct0 + t)*16 + mcol;
            #pragma unroll
            for (int r = 0; r < 8; ++r){
                int kk = k0 + kt*16 + r + half*8;
                atomicAdd(&R[kk*LDR + cbase], acc[kt*4+t][r]);
            }
        }
    }
    if (wv == 0 && mcol == 0){
        #pragma unroll
        for (int kt = 0; kt < 2; ++kt){
            #pragma unroll
            for (int r = 0; r < 8; ++r){
                int kk = k0 + kt*16 + r + half*8;
                atomicAdd(&R[kk*LDR + 512], accS[kt][r]);
            }
        }
    }
}

__global__ void finalize(const float* __restrict__ R, float* __restrict__ out){
    int i = blockIdx.x*blockDim.x + threadIdx.x;
    if (i >= NCLS*NCH) return;
    int k = i >> 9;       // /512
    int c = i & 511;
    float s = R[k*LDR + 512];
    out[i] = R[k*LDR + c] / fmaxf(s, 1e-6f);
}

extern "C" void kernel_launch(void* const* d_in, const int* in_sizes, int n_in,
                              void* d_out, int out_size, void* d_ws, size_t ws_size,
                              hipStream_t stream){
    (void)in_sizes; (void)n_in; (void)out_size; (void)ws_size;
    const float* feats  = (const float*)d_in[0];   // (512, 512*512)
    const float* logits = (const float*)d_in[1];   // (150, 512*512)
    float* wsf  = (float*)d_ws;
    float* sm_m = wsf;                // NPIX floats
    float* sm_r = wsf + NPIX;         // NPIX floats
    float* R    = wsf + 2*NPIX;       // 160*LDR floats

    zeroR<<<(160*LDR + 255)/256, 256, 0, stream>>>(R);
    softmax_stats<<<NPIX/256, 256, 0, stream>>>(logits, sm_m, sm_r);
    dim3 grid(KT2, NCHUNK);
    gemm_wmma<<<grid, 256, 0, stream>>>(feats, logits, sm_m, sm_r, R);
    finalize<<<(NCLS*NCH + 255)/256, 256, 0, stream>>>(R, (float*)d_out);
}
